// MemoryUnit_51522427682836
// MI455X (gfx1250) — matrix operand order repaired and checked by reading, not compiled, not run on hardware
//
#include <hip/hip_runtime.h>

typedef _Float16 v16h __attribute__((ext_vector_type(16)));
typedef _Float16 v8h  __attribute__((ext_vector_type(8)));
typedef float    v8f  __attribute__((ext_vector_type(8)));
typedef float    v4f  __attribute__((ext_vector_type(4)));

#define N_ROWS 131072
#define MDIM   2048
#define ZDIM   128
#define LAMBDA  0.005f
#define EPS_SHRINK 1e-12f
// x is pre-scaled by 2*log2(e): WMMA then yields log2-domain logits, so
// e = exp2(acc) needs no per-element multiply (softmax temp folded in too).
#define XSCALE 2.8853900817779268f

#define ESTRIDE 2056       // bf16 e-buffer row stride (2048 + pad, bank-tuned)
#define SSTRIDE 72         // f16 s-tile row stride (64 + pad, bank-tuned)

// float -> bf16 by truncation (1 VALU op). The downward bias is a uniform
// multiplicative perturbation on e and cancels exactly in sum(t*mem)/sum(t).
__device__ __forceinline__ unsigned short f32_to_bf16_trunc(float f) {
  union { float f; unsigned u; } q; q.f = f;
  return (unsigned short)(q.u >> 16);
}
__device__ __forceinline__ float bf16_lo(unsigned u) {   // low 16 bits -> f32
  union { unsigned u; float f; } q; q.u = u << 16; return q.f;
}
__device__ __forceinline__ float bf16_hi(unsigned u) {   // high 16 bits -> f32
  union { unsigned u; float f; } q; q.u = u & 0xFFFF0000u; return q.f;
}

// ---------------------------------------------------------------------------
// Prep: fp32 mem [M][Z] -> f16 memh [M][Z] and f16 memT [Z][M] (1 MB, L2-resident)
// ---------------------------------------------------------------------------
__global__ void prep_mem_kernel(const float* __restrict__ mem,
                                _Float16* __restrict__ memh,
                                _Float16* __restrict__ memT) {
  int idx = blockIdx.x * blockDim.x + threadIdx.x;
  if (idx >= MDIM * ZDIM) return;
  int m = idx / ZDIM, z = idx % ZDIM;
  _Float16 h = (_Float16)mem[idx];
  memh[idx] = h;
  memT[z * MDIM + m] = h;
}

// ---------------------------------------------------------------------------
// Fused memory-unit. One WG = 256 threads (8 waves) handles 32 rows of x.
//   wave w: row block rb = w&1 (16 rows), col/z block cb = w>>1 (4 blocks)
// Pass 1: att GEMM (WMMA f16, log2-scaled) -> e = exp2(acc) stored bf16 in
//         LDS (128.5 KB), Z = sum(e). No max-subtraction (logits bounded).
// Pass 2: shrink in e-space (all 1/Z factors cancel algebraically),
//         f16 s-tile (double-buffered, 1 barrier/chunk) -> out GEMM (WMMA).
// att never touches HBM; exactly one transcendental per attention element.
// ---------------------------------------------------------------------------
__global__ __launch_bounds__(256)
void memunit_fused_kernel(const float* __restrict__ x,
                          const _Float16* __restrict__ memh,
                          const _Float16* __restrict__ memT,
                          float* __restrict__ out) {
  __shared__ unsigned short ebuf[32 * ESTRIDE];   // 128.5 KB  e = exp2(acc), bf16
  __shared__ _Float16 sbuf[2][32 * SSTRIDE];      // 9 KB      double-buffered s-tile
  __shared__ float s_z[4][2][16];                 // per (cb, rb, row) partial Z
  __shared__ float s_l1[32];                      // per-row L1 (e-space)

  const int tid  = threadIdx.x;
  const int wave = tid >> 5;
  const int lane = tid & 31;
  const int c  = lane & 15;          // column-in-tile
  const int h  = lane >> 4;          // lane half (WMMA layout select)
  const int rb = wave & 1;           // row block (16 rows)
  const int cb = wave >> 1;          // GEMM1 col block / GEMM2 z block (0..3)

  const int n0 = blockIdx.x * 32;

  // ---- A fragments: 16 rows of x * XSCALE (f32 -> f16), in VGPRs for good.
  // k order used here for A (B matches): elems 0..15 -> K = h*16+e
  v16h afrag[4];
  {
    const float* xr = x + (size_t)(n0 + rb * 16 + c) * ZDIM;
#pragma unroll
    for (int kc = 0; kc < 4; ++kc) {
      const float* p0 = xr + kc * 32 + h * 16;
      v4f a0 = *(const v4f*)(p0);
      v4f a1 = *(const v4f*)(p0 + 4);
      v4f b0 = *(const v4f*)(p0 + 8);
      v4f b1 = *(const v4f*)(p0 + 12);
      v16h f;
#pragma unroll
      for (int e = 0; e < 4; ++e) {
        f[e]      = (_Float16)(a0[e] * XSCALE);
        f[e + 4]  = (_Float16)(a1[e] * XSCALE);
        f[e + 8]  = (_Float16)(b0[e] * XSCALE);
        f[e + 12] = (_Float16)(b1[e] * XSCALE);
      }
      afrag[kc] = f;
    }
  }

  // ======================= PASS 1: GEMM1 + exp2 + Z ========================
  float rsum[8];
#pragma unroll
  for (int i = 0; i < 8; ++i) rsum[i] = 0.0f;

  for (int mc = 0; mc < MDIM / 64; ++mc) {
    const int mbase = mc * 64;
    const _Float16* brow = memh + (size_t)(mbase + cb * 16 + c) * ZDIM;
    __builtin_prefetch(brow + 64 * ZDIM, 0, 1);   // warm next chunk's B rows
    v8f acc = {};
#pragma unroll
    for (int kc = 0; kc < 4; ++kc) {
      v16h bfrag = *(const v16h*)(brow + kc * 32 + h * 16);
      acc = __builtin_amdgcn_wmma_f32_16x16x32_f16(false, afrag[kc], false, bfrag,
                                                   (short)0, acc, false, false);
    }
    // C layout: vgpr i, lane (c,h) -> row rb*16+i+8h, col mbase+cb*16+c
    const int ecol = mbase + cb * 16 + c;
#pragma unroll
    for (int i = 0; i < 8; ++i) {
      float e = __builtin_amdgcn_exp2f(acc[i]);   // acc already log2-scaled
      rsum[i] += e;
      ebuf[(rb * 16 + i + 8 * h) * ESTRIDE + ecol] = f32_to_bf16_trunc(e);
    }
  }
  // sum across the 16 lanes of each half, then across the 4 column-waves
#pragma unroll
  for (int off = 1; off < 16; off <<= 1) {
#pragma unroll
    for (int i = 0; i < 8; ++i)
      rsum[i] += __shfl_xor(rsum[i], off, 32);
  }
  if (c == 0) {
#pragma unroll
    for (int i = 0; i < 8; ++i)
      s_z[cb][rb][i + 8 * h] = rsum[i];
  }
  if (tid < 32) s_l1[tid] = 0.0f;
  __syncthreads();

  // ================= PASS 2: shrink (e-space) + GEMM2 ======================
  // s_p = e*relu(e-lam*Z) / (Z*(|e-lam*Z|+eps*Z)); every Z cancels in
  // out = (sum t_e * mem) / (sum t_e) with t_e = e*relu(e-C)/(|e-C|+Ce).
  const int r = tid & 31;            // elementwise stage: row per thread
  const int g = tid >> 5;            // elementwise stage: 8-wide m group
  float Cr, Cer;
  {
    float Z = s_z[0][r >> 4][r & 15] + s_z[1][r >> 4][r & 15] +
              s_z[2][r >> 4][r & 15] + s_z[3][r >> 4][r & 15];
    Cr  = LAMBDA * Z;
    Cer = EPS_SHRINK * Z;
  }

  float l1p = 0.0f;
  v8f oacc[2] = {};
  const unsigned short* erow = ebuf + r * ESTRIDE;

  for (int mc = 0; mc < MDIM / 64; ++mc) {
    const int mbase = mc * 64;
    const int par = mc & 1;

    // ---- stage 1: all 256 threads, 8 elements each (contiguous b128 LDS I/O)
    {
      uint4 ev = *(const uint4*)(erow + mbase + g * 8);
      float e[8];
      e[0] = bf16_lo(ev.x); e[1] = bf16_hi(ev.x);
      e[2] = bf16_lo(ev.y); e[3] = bf16_hi(ev.y);
      e[4] = bf16_lo(ev.z); e[5] = bf16_hi(ev.z);
      e[6] = bf16_lo(ev.w); e[7] = bf16_hi(ev.w);
      v8h sv;
#pragma unroll
      for (int j = 0; j < 8; ++j) {
        float d = e[j] - Cr;
        float t = fmaxf(d, 0.0f) * e[j] * __builtin_amdgcn_rcpf(fabsf(d) + Cer);
        l1p += t;
        sv[j] = (_Float16)t;
      }
      *(v8h*)(&sbuf[par][r * SSTRIDE + g * 8]) = sv;
    }
    __syncthreads();

    // ---- stage 2: out[16x32] += S[16x64] * mem[64x32] (4 WMMA / wave)
    const _Float16* sr = &sbuf[par][(rb * 16 + c) * SSTRIDE];
#pragma unroll
    for (int k2 = 0; k2 < 2; ++k2) {
      v8h lo = *(const v8h*)(sr + k2 * 32 + h * 16);
      v8h hi = *(const v8h*)(sr + k2 * 32 + h * 16 + 8);
      v16h a2 = __builtin_shufflevector(lo, hi, 0, 1, 2, 3, 4, 5, 6, 7,
                                        8, 9, 10, 11, 12, 13, 14, 15);
#pragma unroll
      for (int t = 0; t < 2; ++t) {
        const _Float16* btr = memT + (size_t)(cb * 32 + t * 16 + c) * MDIM
                              + mbase + k2 * 32 + h * 16;
        v16h b2 = *(const v16h*)btr;
        oacc[t] = __builtin_amdgcn_wmma_f32_16x16x32_f16(false, a2, false, b2,
                                                         (short)0, oacc[t], false, false);
      }
    }
    // double-buffered sbuf: next iteration writes the other half -> one barrier
  }

  // ---- per-row L1 (e-space): 8 partial owners per row -> LDS atomics
  atomicAdd(&s_l1[r], l1p);
  __syncthreads();

  // ---- epilogue: out = oacc / l1  (guard only for the all-shrunk-to-zero row)
#pragma unroll
  for (int i = 0; i < 8; ++i) {
    const int row = rb * 16 + i + 8 * h;
    const float inv = __builtin_amdgcn_rcpf(fmaxf(s_l1[row], 1e-30f));
    float* orow = out + (size_t)(n0 + row) * ZDIM + cb * 32 + c;
    orow[0]  = oacc[0][i] * inv;
    orow[16] = oacc[1][i] * inv;
  }
}

// ---------------------------------------------------------------------------
extern "C" void kernel_launch(void* const* d_in, const int* in_sizes, int n_in,
                              void* d_out, int out_size, void* d_ws, size_t ws_size,
                              hipStream_t stream) {
  const float* x   = (const float*)d_in[0];   // [131072, 128] f32
  const float* mem = (const float*)d_in[1];   // [2048, 128]  f32
  float* out = (float*)d_out;                 // [131072, 128] f32

  _Float16* memh = (_Float16*)d_ws;           // 512 KB
  _Float16* memT = memh + MDIM * ZDIM;        // 512 KB

  prep_mem_kernel<<<(MDIM * ZDIM + 255) / 256, 256, 0, stream>>>(mem, memh, memT);
  memunit_fused_kernel<<<N_ROWS / 32, 256, 0, stream>>>(x, memh, memT, out);
}
